// AskeyWilsonKANLayer_30331059044960
// MI455X (gfx1250) — matrix operand order repaired and checked by reading, not compiled, not run on hardware
//
#include <hip/hip_runtime.h>
#include <hip/hip_bf16.h>

// ---- types ----
typedef __attribute__((ext_vector_type(4)))  __bf16 v4bf;
typedef __attribute__((ext_vector_type(8)))  __bf16 v8bf;
typedef __attribute__((ext_vector_type(16))) __bf16 v16bf;
typedef __attribute__((ext_vector_type(8)))  float  v8f;
typedef __attribute__((ext_vector_type(4)))  float  f32x4;                // 16B-aligned
typedef __attribute__((ext_vector_type(4), aligned(4))) float f32x4u;     // 4B-aligned (coeffs groups)

#define DEG 8
#define ND  9            // degree + 1
#define BM  128          // C tile rows (batch)  -- small side: recurrence-heavy A fill
#define BN  256          // C tile cols (output) -- large side: cheap load-only B fill
#define KI  32           // K step per WMMA (i-chunk)
#define LDA (KI + 8)     // padded LDS row stride: 80B rows, conflict-free b128, 8B-aligned b64
#define INPUT_DIM  1024
#define OUTPUT_DIM 1024
#define BATCH      8192
#define NCHUNK (INPUT_DIM / KI)

__global__ __launch_bounds__(256)
void aw_kan_wmma(const float* __restrict__ x,
                 const float* __restrict__ pa, const float* __restrict__ pb,
                 const float* __restrict__ pc, const float* __restrict__ pd,
                 const float* __restrict__ pq,
                 const float* __restrict__ coeffs,
                 float* __restrict__ out)
{
    extern __shared__ __align__(16) char smem[];
    __bf16* As = (__bf16*)smem;                                   // [ND][BM][LDA]
    __bf16* Bs = (__bf16*)(smem + (size_t)ND * BM * LDA * 2);     // [ND][BN][LDA] (N-major)

    const int tid   = threadIdx.x;
    const int lane  = tid & 31;
    const int wave  = tid >> 5;       // 0..7
    const int waveM = wave >> 2;      // 0..1 -> 64-row band
    const int waveN = wave & 3;       // 0..3 -> 64-col band
    const int mBase = blockIdx.y * BM;
    const int nBase = blockIdx.x * BN;

    // ---- scalar params & folded recurrence coefficients ----
    const float sa = pa[0], sb = pb[0], sc = pc[0], sd = pd[0], sq = pq[0];
    const float abcd = sa * sb * sc * sd;
    const float invP1 = 1.f / (1.f + abcd * sq * sq);
    const float c1 = 2.f * (1.f + sa * sb * sq) * invP1;
    const float c0 = -(sa + sb) * (1.f + sc * sd * sq) * invP1;

    float In[ND], A2[ND], C2[ND];     // In = 1/(1-q^n), A2 = An*In, C2 = Cn*In
    {
        float qn1  = sq;        // q^(n-1)
        float q2n2 = sq * sq;   // q^(2n-2)
        #pragma unroll
        for (int n = 2; n <= DEG; ++n) {
            float qn   = qn1 * sq;
            float q2n1 = q2n2 * sq;
            float q2n  = q2n1 * sq;
            float t1 = (1.f - sa * sb * qn1) * (1.f - sc * sd * qn1) * (1.f - abcd * q2n2);
            float An = t1 / ((1.f - abcd * q2n1) * (1.f - abcd * q2n));
            float Cn = ((1.f - qn) * t1) / ((1.f - abcd * q2n2) * (1.f - abcd * q2n1));
            In[n] = 1.f / (1.f - qn);
            A2[n] = An * In[n];
            C2[n] = Cn * In[n];
            qn1  = qn;
            q2n2 = q2n;
        }
    }

    v8f acc[4][4];
    #pragma unroll
    for (int tm = 0; tm < 4; ++tm)
        #pragma unroll
        for (int tn = 0; tn < 4; ++tn)
            acc[tm][tn] = (v8f){0.f, 0.f, 0.f, 0.f, 0.f, 0.f, 0.f, 0.f};

    const int mRow  = lane & 15;
    const int kHalf = lane >> 4;

    // ---- A-fill thread geometry (4 groups of 4 consecutive k per thread) ----
    const int gm[4]  = { (0 * 256 + tid) >> 3, (1 * 256 + tid) >> 3,
                         (2 * 256 + tid) >> 3, (3 * 256 + tid) >> 3 };
    const int gk4    = (tid & 7) * 4;

    // degree-0 slab of A is all-ones for every chunk: write it once
    {
        const v4bf ones = (v4bf){(__bf16)1.f, (__bf16)1.f, (__bf16)1.f, (__bf16)1.f};
        #pragma unroll
        for (int j = 0; j < 4; ++j)
            *(v4bf*)(As + (size_t)gm[j] * LDA + gk4) = ones;
    }

    // prefetch chunk 0 x tile into registers
    f32x4 xreg[4];
    #pragma unroll
    for (int j = 0; j < 4; ++j)
        xreg[j] = *(const f32x4*)(x + (size_t)(mBase + gm[j]) * INPUT_DIM + gk4);

    for (int ic = 0; ic < NCHUNK; ++ic) {
        // ---- A: 4 parallel recurrences per group from prefetched x, packed b64 LDS stores ----
        #pragma unroll
        for (int j = 0; j < 4; ++j) {
            const f32x4 xv = xreg[j];
            __bf16* ap = As + (size_t)gm[j] * LDA + gk4;

            float pm2[4], pm1[4];
            #pragma unroll
            for (int s = 0; s < 4; ++s) { pm2[s] = 1.f; pm1[s] = c1 * xv[s] + c0; }
            *(v4bf*)(ap + BM * LDA) =
                (v4bf){(__bf16)pm1[0], (__bf16)pm1[1], (__bf16)pm1[2], (__bf16)pm1[3]};
            #pragma unroll
            for (int n = 2; n <= DEG; ++n) {
                float pn[4];
                #pragma unroll
                for (int s = 0; s < 4; ++s) {
                    // pn = ((2x - An)p1 - Cn p2) / (1-q^n), folded to 3 FMA-class ops:
                    const float r = __builtin_fmaf(2.f * xv[s], In[n], -A2[n]);
                    const float t = C2[n] * pm2[s];
                    pn[s] = __builtin_fmaf(pm1[s], r, -t);
                    pm2[s] = pm1[s];
                    pm1[s] = pn[s];
                }
                *(v4bf*)(ap + (size_t)n * BM * LDA) =
                    (v4bf){(__bf16)pn[0], (__bf16)pn[1], (__bf16)pn[2], (__bf16)pn[3]};
            }
        }

        // ---- B: coeffs[i, o, 0..8] contiguous per (i,o); 4 i-rows per thread-group,
        //         repack per-degree into packed b64 stores to Bs[d][o][i] ----
        #pragma unroll
        for (int j = 0; j < ((KI / 4) * BN) / 256; ++j) {   // 8 groups / thread
            const int g  = j * 256 + tid;
            const int o  = g & (BN - 1);                    // o fastest -> coalesced-ish
            const int i4 = (g >> 8) * 4;
            float v[4][ND];
            #pragma unroll
            for (int s = 0; s < 4; ++s) {
                const float* gp = coeffs +
                    ((size_t)(ic * KI + i4 + s) * OUTPUT_DIM + (nBase + o)) * ND;
                f32x4u t0 = *(const f32x4u*)(gp);
                f32x4u t1 = *(const f32x4u*)(gp + 4);
                v[s][0] = t0.x; v[s][1] = t0.y; v[s][2] = t0.z; v[s][3] = t0.w;
                v[s][4] = t1.x; v[s][5] = t1.y; v[s][6] = t1.z; v[s][7] = t1.w;
                v[s][8] = gp[8];
            }
            __bf16* bp = Bs + (size_t)o * LDA + i4;
            #pragma unroll
            for (int dd = 0; dd < ND; ++dd) {
                *(v4bf*)(bp + (size_t)dd * BN * LDA) =
                    (v4bf){(__bf16)v[0][dd], (__bf16)v[1][dd],
                           (__bf16)v[2][dd], (__bf16)v[3][dd]};
            }
        }

        __syncthreads();

        // ---- prefetch next chunk's x while WMMA runs (latency hidden under XDL) ----
        if (ic + 1 < NCHUNK) {
            #pragma unroll
            for (int j = 0; j < 4; ++j)
                xreg[j] = *(const f32x4*)(x + (size_t)(mBase + gm[j]) * INPUT_DIM
                                            + (ic + 1) * KI + gk4);
        }

        // ---- WMMA: 9 degree slabs x (4 M-tiles x 4 N-tiles) per wave = 144 per chunk ----
        #pragma unroll
        for (int dd = 0; dd < ND; ++dd) {
            const __bf16* aSlab = As + (size_t)dd * BM * LDA;
            const __bf16* bSlab = Bs + (size_t)dd * BN * LDA;

            v16bf afrag[4];
            #pragma unroll
            for (int tm = 0; tm < 4; ++tm) {
                // A 16x32 bf16: lanes0-15 rows M0-15 hold K[kHalf*8..+7] and K[16+kHalf*8..+7]
                const __bf16* ar = aSlab +
                    (size_t)(waveM * 64 + tm * 16 + mRow) * LDA + kHalf * 8;
                v8bf lo = *(const v8bf*)(ar);
                v8bf hi = *(const v8bf*)(ar + 16);
                afrag[tm] = __builtin_shufflevector(lo, hi,
                    0, 1, 2, 3, 4, 5, 6, 7, 8, 9, 10, 11, 12, 13, 14, 15);
            }

            #pragma unroll
            for (int tn = 0; tn < 4; ++tn) {
                // B 32x16 bf16: lane = column n, K[kHalf*8..+7] and K[16+kHalf*8..+7]
                const __bf16* br = bSlab +
                    (size_t)(waveN * 64 + tn * 16 + mRow) * LDA + kHalf * 8;
                v8bf blo = *(const v8bf*)(br);
                v8bf bhi = *(const v8bf*)(br + 16);
                v16bf bfrag = __builtin_shufflevector(blo, bhi,
                    0, 1, 2, 3, 4, 5, 6, 7, 8, 9, 10, 11, 12, 13, 14, 15);
                #pragma unroll
                for (int tm = 0; tm < 4; ++tm) {
                    acc[tm][tn] = __builtin_amdgcn_wmma_f32_16x16x32_bf16(
                        false, afrag[tm], false, bfrag,
                        (short)0, acc[tm][tn], false, false);
                }
            }
        }

        __syncthreads();
    }

    // ---- store C: VGPR r of 16x16 f32 D holds (M = r + 8*kHalf, N = lane&15) ----
    const int nCol = lane & 15;
    #pragma unroll
    for (int tm = 0; tm < 4; ++tm) {
        #pragma unroll
        for (int tn = 0; tn < 4; ++tn) {
            float* orow = out +
                (size_t)(mBase + waveM * 64 + tm * 16 + kHalf * 8) * OUTPUT_DIM +
                (nBase + waveN * 64 + tn * 16 + nCol);
            #pragma unroll
            for (int r = 0; r < 8; ++r)
                orow[(size_t)r * OUTPUT_DIM] = acc[tm][tn][r];
        }
    }
}

extern "C" void kernel_launch(void* const* d_in, const int* in_sizes, int n_in,
                              void* d_out, int out_size, void* d_ws, size_t ws_size,
                              hipStream_t stream) {
    const float* x      = (const float*)d_in[0];
    const float* a      = (const float*)d_in[1];
    const float* b      = (const float*)d_in[2];
    const float* c      = (const float*)d_in[3];
    const float* d      = (const float*)d_in[4];
    const float* q      = (const float*)d_in[5];
    const float* coeffs = (const float*)d_in[6];
    float* out = (float*)d_out;

    const size_t smem = (size_t)ND * (BM + BN) * LDA * sizeof(__bf16); // 276,480 B (< 320KB/WGP)
    (void)hipFuncSetAttribute((const void*)aw_kan_wmma,
                              hipFuncAttributeMaxDynamicSharedMemorySize,
                              (int)smem);

    dim3 grid(OUTPUT_DIM / BN, BATCH / BM);   // (4, 64)
    dim3 block(256);                           // 8 wave32s
    aw_kan_wmma<<<grid, block, smem, stream>>>(x, a, b, c, d, q, coeffs, out);
}
